// Encoder_38302518345840
// MI455X (gfx1250) — matrix operand order repaired and checked by reading, not compiled, not run on hardware
//
#include <hip/hip_runtime.h>
#include <math.h>

// ---------- problem constants (match reference) ----------
#define BB   4
#define SS   2048
#define DD   1024
#define DFFN 4096
#define LNEPS 1e-5f

// ---------- WMMA vector types ----------
typedef __attribute__((ext_vector_type(16))) __bf16        v16bf;
typedef __attribute__((ext_vector_type(8)))  float         v8f;
typedef __attribute__((ext_vector_type(8)))  unsigned int  v8u;

// float -> bf16 (round-to-nearest-even)
__device__ __forceinline__ unsigned short f2bf(float f) {
  unsigned int u = __builtin_bit_cast(unsigned int, f);
  u += 0x7FFFu + ((u >> 16) & 1u);
  return (unsigned short)(u >> 16);
}

// A fragment: per-lane 16 bf16 = K[koff..koff+7] and K[koff+16..koff+23] (two b128s, 32B apart)
__device__ __forceinline__ v16bf load_frag_a(const unsigned short* base) {
  const uint4* p = reinterpret_cast<const uint4*>(base);
  uint4 lo = p[0];
  uint4 hi = p[2];
  v8u u;
  u[0]=lo.x; u[1]=lo.y; u[2]=lo.z; u[3]=lo.w;
  u[4]=hi.x; u[5]=hi.y; u[6]=hi.z; u[7]=hi.w;
  return __builtin_bit_cast(v16bf, u);
}

// B fragment from LDS: per-lane 16 bf16 = K[koff..koff+7] and K[koff+16..koff+23] (two b128s, 32B apart)
__device__ __forceinline__ v16bf load_frag_b(const unsigned short* base) {
  const uint4* p = reinterpret_cast<const uint4*>(base);
  uint4 lo = p[0];
  uint4 hi = p[2];
  v8u u;
  u[0]=lo.x; u[1]=lo.y; u[2]=lo.z; u[3]=lo.w;
  u[4]=hi.x; u[5]=hi.y; u[6]=hi.z; u[7]=hi.w;
  return __builtin_bit_cast(v16bf, u);
}

// ---- register-staged B tile (32K x 64N), LDS layout bt[n][k] (bf16) ----
struct StageRegs { uint4 lo, hi; };

// BCOLMAJOR==1: B(k,n) = Bm[n*ldb + k]  (K-contiguous; pure b128 copies)
// BCOLMAJOR==0: B(k,n) = Bm[k*ldb + n]  (N-contiguous; vector load, b16 scatter)
template <int BCOLMAJOR>
__device__ __forceinline__ StageRegs load_b_tile_regs(const unsigned short* __restrict__ Bm,
                                                      long ldb, int k0, int n0, int t) {
  StageRegs r;
  if (BCOLMAJOR) {
    const int nl = t >> 1;          // 0..63
    const int kb = (t & 1) * 16;    // 0 or 16
    const uint4* p = reinterpret_cast<const uint4*>(Bm + (size_t)(n0 + nl) * (size_t)ldb + k0 + kb);
    r.lo = p[0];
    r.hi = p[1];
  } else {
    const int kk  = t >> 2;         // 0..31
    const int nnb = (t & 3) * 16;   // 0,16,32,48
    const uint4* p = reinterpret_cast<const uint4*>(Bm + (size_t)(k0 + kk) * (size_t)ldb + n0 + nnb);
    r.lo = p[0];
    r.hi = p[1];
  }
  return r;
}

template <int BCOLMAJOR>
__device__ __forceinline__ void store_b_tile_lds(unsigned short* __restrict__ dst,
                                                 StageRegs r, int t) {
  if (BCOLMAJOR) {
    const int nl = t >> 1;
    const int kb = (t & 1) * 16;
    uint4* q = reinterpret_cast<uint4*>(dst + nl * 32 + kb);
    q[0] = r.lo;
    q[1] = r.hi;
  } else {
    const int kk  = t >> 2;
    const int nnb = (t & 3) * 16;
    unsigned int wds[8];
    wds[0]=r.lo.x; wds[1]=r.lo.y; wds[2]=r.lo.z; wds[3]=r.lo.w;
    wds[4]=r.hi.x; wds[5]=r.hi.y; wds[6]=r.hi.z; wds[7]=r.hi.w;
#pragma unroll
    for (int j = 0; j < 16; ++j) {
      dst[(nnb + j) * 32 + kk] = (unsigned short)(wds[j >> 1] >> ((j & 1) * 16));
    }
  }
}

// ---------------------------------------------------------------------------
// bf16 WMMA GEMM:  C = epilogue( A[M,K] (row-major bf16) * B * scale + evec )
// Block = 128 threads (4 waves). C tile = 128(M) x 64(N); wave w owns rows
// [m0+32w, m0+32w+32) as two 16-row strips x 4 N-tiles = 8 accumulators.
// K loop manually unrolled 2x with ping-ponged register sets + LDS buffers:
// no pipeline-rotation movs, one barrier per 32-K step, global fetches for
// step i+1 issue before the WMMA burst of step i. Requires K % 64 == 0.
// GELU_BF16 == 1: exact GELU then store bf16; else store f32.
// ---------------------------------------------------------------------------
template <int GELU_BF16, int BCOLMAJOR>
__global__ __launch_bounds__(128)
void gemm_wmma_bf16(const unsigned short* __restrict__ A,
                    const unsigned short* __restrict__ Bm,
                    void* __restrict__ Cout,
                    const float* __restrict__ evec,
                    float scale,
                    int M, int N, int K, long ldb,
                    long bsA, long bsB, long bsC, long bsE) {
  __shared__ unsigned short bt[2][64 * 32]; // ping-pong transposed B tile, 2 x 4 KB

  const int batch = blockIdx.z;
  A  += (size_t)batch * (size_t)bsA;
  Bm += (size_t)batch * (size_t)bsB;
  const float* ev = evec ? (evec + (size_t)batch * (size_t)bsE) : nullptr;

  const int m0 = blockIdx.y * 128;
  const int n0 = blockIdx.x * 64;
  const int t    = threadIdx.x;
  const int lane = t & 31;
  const int wave = t >> 5;
  const int l16   = lane & 15;
  const int lhalf = lane >> 4;

  v8f acc[2][4] = {};

  const unsigned short* arowp0 = A + (size_t)(m0 + wave * 32 + l16)      * (size_t)K;
  const unsigned short* arowp1 = A + (size_t)(m0 + wave * 32 + 16 + l16) * (size_t)K;
  const int aoff = lhalf * 8;
  const int boff = (l16 * 32) + lhalf * 8; // lane's B-fragment base within a tile

  // ---- prologue: stage tile 0 into LDS buf0, A frags (set0) for k=0 ----
  {
    StageRegs sr = load_b_tile_regs<BCOLMAJOR>(Bm, ldb, 0, n0, t);
    store_b_tile_lds<BCOLMAJOR>(bt[0], sr, t);
  }
  v16bf a0_0 = load_frag_a(arowp0 + aoff);
  v16bf a1_0 = load_frag_a(arowp1 + aoff);
  v16bf a0_1, a1_1;
  __syncthreads();

  for (int k0 = 0; k0 < K; k0 += 64) {
    // ---------------- half-step A: compute k0 from buf0/set0 ----------------
    {
      // fetch k0+32 (always exists: K % 64 == 0)
      StageRegs nr = load_b_tile_regs<BCOLMAJOR>(Bm, ldb, k0 + 32, n0, t);
      a0_1 = load_frag_a(arowp0 + k0 + 32 + aoff);
      a1_1 = load_frag_a(arowp1 + k0 + 32 + aoff);
      __builtin_prefetch(arowp0 + k0 + 96, 0, 1); // global_prefetch_b8
      __builtin_prefetch(arowp1 + k0 + 96, 0, 1);

      v16bf bfr[4];
#pragma unroll
      for (int nt = 0; nt < 4; ++nt)
        bfr[nt] = load_frag_b(&bt[0][nt * 16 * 32 + boff]);
#pragma unroll
      for (int nt = 0; nt < 4; ++nt) {
        acc[0][nt] = __builtin_amdgcn_wmma_f32_16x16x32_bf16(
            false, a0_0, false, bfr[nt], (short)0, acc[0][nt], false, false);
        acc[1][nt] = __builtin_amdgcn_wmma_f32_16x16x32_bf16(
            false, a1_0, false, bfr[nt], (short)0, acc[1][nt], false, false);
      }
      store_b_tile_lds<BCOLMAJOR>(bt[1], nr, t);
      __syncthreads();
    }

    // ---------------- half-step B: compute k0+32 from buf1/set1 ----------------
    {
      const bool more = (k0 + 64) < K;
      StageRegs nr;
      if (more) {
        nr = load_b_tile_regs<BCOLMAJOR>(Bm, ldb, k0 + 64, n0, t);
        a0_0 = load_frag_a(arowp0 + k0 + 64 + aoff);
        a1_0 = load_frag_a(arowp1 + k0 + 64 + aoff);
        __builtin_prefetch(arowp0 + k0 + 128, 0, 1);
        __builtin_prefetch(arowp1 + k0 + 128, 0, 1);
      }

      v16bf bfr[4];
#pragma unroll
      for (int nt = 0; nt < 4; ++nt)
        bfr[nt] = load_frag_b(&bt[1][nt * 16 * 32 + boff]);
#pragma unroll
      for (int nt = 0; nt < 4; ++nt) {
        acc[0][nt] = __builtin_amdgcn_wmma_f32_16x16x32_bf16(
            false, a0_1, false, bfr[nt], (short)0, acc[0][nt], false, false);
        acc[1][nt] = __builtin_amdgcn_wmma_f32_16x16x32_bf16(
            false, a1_1, false, bfr[nt], (short)0, acc[1][nt], false, false);
      }
      if (more) store_b_tile_lds<BCOLMAJOR>(bt[0], nr, t);
      __syncthreads();
    }
  }

  // epilogue; C/D layout: VGPR i -> row = i + 8*(lane>=16), col = lane&15
#pragma unroll
  for (int mi = 0; mi < 2; ++mi) {
#pragma unroll
    for (int nt = 0; nt < 4; ++nt) {
      const int col = n0 + nt * 16 + l16;
      const float e = ev ? ev[col] : 0.0f;
#pragma unroll
      for (int i = 0; i < 8; ++i) {
        const int row = m0 + wave * 32 + mi * 16 + lhalf * 8 + i;
        float v = acc[mi][nt][i] * scale + e;
        const size_t idx = (size_t)batch * (size_t)bsC + (size_t)row * (size_t)N + (size_t)col;
        if (GELU_BF16) {
          v = 0.5f * v * (1.0f + erff(v * 0.70710678118654752f)); // exact GELU
          ((unsigned short*)Cout)[idx] = f2bf(v);
        } else {
          ((float*)Cout)[idx] = v;
        }
      }
    }
  }
}

// ---------------- f32 -> bf16 convert ----------------
__global__ __launch_bounds__(256)
void cvt_to_bf16(const float* __restrict__ in, unsigned short* __restrict__ out, long n) {
  long i = (long)blockIdx.x * blockDim.x + threadIdx.x;
  const long stride = (long)gridDim.x * blockDim.x;
  for (; i < n; i += stride) out[i] = f2bf(in[i]);
}

// ---------------- row softmax: scores f32[S] -> attn bf16[S] ----------------
__global__ __launch_bounds__(256)
void softmax_rows(const float* __restrict__ s, unsigned short* __restrict__ attn, int S) {
  const long row = blockIdx.x;
  const float* sr = s + row * (long)S;
  unsigned short* ar = attn + row * (long)S;
  const int t = threadIdx.x;
  __shared__ float red[8];

  float lmax = -3.4e38f;
  for (int j = t; j < S; j += 256) lmax = fmaxf(lmax, sr[j]);
  for (int o = 16; o > 0; o >>= 1) lmax = fmaxf(lmax, __shfl_xor(lmax, o, 32));
  if ((t & 31) == 0) red[t >> 5] = lmax;
  __syncthreads();
  float m = red[0];
#pragma unroll
  for (int w = 1; w < 8; ++w) m = fmaxf(m, red[w]);

  float lsum = 0.0f;
  for (int j = t; j < S; j += 256) lsum += __expf(sr[j] - m);
  for (int o = 16; o > 0; o >>= 1) lsum += __shfl_xor(lsum, o, 32);
  __syncthreads();
  if ((t & 31) == 0) red[t >> 5] = lsum;
  __syncthreads();
  float sum = 0.0f;
#pragma unroll
  for (int w = 0; w < 8; ++w) sum += red[w];
  const float inv = 1.0f / sum;

  for (int j = t; j < S; j += 256) ar[j] = f2bf(__expf(sr[j] - m) * inv);
}

// ---------------- residual add + LayerNorm (optionally emit bf16 copy) ----------------
__global__ __launch_bounds__(256)
void add_layernorm(const float* __restrict__ x, const float* __restrict__ y,
                   const float* __restrict__ g, const float* __restrict__ b,
                   float* __restrict__ out, unsigned short* __restrict__ out_bf) {
  const long row = blockIdx.x;
  const float* xr = x + row * (long)DD;
  const float* yr = y + row * (long)DD;
  const int t = threadIdx.x;
  __shared__ float rs[8], rs2[8];

  float vals[DD / 256];
  float s = 0.0f, s2 = 0.0f;
#pragma unroll
  for (int c = 0; c < DD / 256; ++c) {
    const int j = t + c * 256;
    const float v = xr[j] + yr[j];
    vals[c] = v;
    s += v;
    s2 += v * v;
  }
  for (int o = 16; o > 0; o >>= 1) {
    s  += __shfl_xor(s, o, 32);
    s2 += __shfl_xor(s2, o, 32);
  }
  if ((t & 31) == 0) { rs[t >> 5] = s; rs2[t >> 5] = s2; }
  __syncthreads();
  s = 0.0f; s2 = 0.0f;
#pragma unroll
  for (int w = 0; w < 8; ++w) { s += rs[w]; s2 += rs2[w]; }

  const float mean = s * (1.0f / DD);
  const float var  = s2 * (1.0f / DD) - mean * mean;
  const float inv  = rsqrtf(var + LNEPS);

#pragma unroll
  for (int c = 0; c < DD / 256; ++c) {
    const int j = t + c * 256;
    const float nv = (vals[c] - mean) * inv * g[j] + b[j];
    out[row * (long)DD + j] = nv;
    if (out_bf) out_bf[row * (long)DD + j] = f2bf(nv);
  }
}

// ---------------------------------------------------------------------------
extern "C" void kernel_launch(void* const* d_in, const int* in_sizes, int n_in,
                              void* d_out, int out_size, void* d_ws, size_t ws_size,
                              hipStream_t stream) {
  const float* x    = (const float*)d_in[0];
  const float* mask = (const float*)d_in[1];
  const float* w1   = (const float*)d_in[2];
  const float* b1   = (const float*)d_in[3];
  const float* w2   = (const float*)d_in[4];
  const float* b2   = (const float*)d_in[5];
  const float* g1   = (const float*)d_in[6];
  const float* be1  = (const float*)d_in[7];
  const float* g2   = (const float*)d_in[8];
  const float* be2  = (const float*)d_in[9];
  float* out = (float*)d_out;

  // ---- workspace map (aliased where lifetimes allow) ----
  char* w = (char*)d_ws;
  size_t off = 0;
  auto take = [&](size_t bytes) -> void* {
    void* p = w + off;
    off += (bytes + 255) & ~(size_t)255;
    return p;
  };
  unsigned short* x_bf   = (unsigned short*)take((size_t)BB * SS * DD * 2);   // 16.8 MB
  unsigned short* w1_bf  = (unsigned short*)take((size_t)DD * DFFN * 2);      //  8.4 MB
  unsigned short* w2_bf  = (unsigned short*)take((size_t)DFFN * DD * 2);      //  8.4 MB
  void*           big    = take((size_t)BB * SS * SS * 4);                    // 67 MB: scores f32, later act bf16
  unsigned short* attn   = (unsigned short*)take((size_t)BB * SS * SS * 2);   // 33.6 MB
  float*          tmpf   = (float*)take((size_t)BB * SS * DD * 4);            // 33.6 MB: attn_out, later ffn
  float*          h      = (float*)take((size_t)BB * SS * DD * 4);            // 33.6 MB
  unsigned short* h_bf   = (unsigned short*)take((size_t)BB * SS * DD * 2);   // 16.8 MB
  float*          scores = (float*)big;
  unsigned short* act    = (unsigned short*)big;

  // 1) bf16 conversions
  cvt_to_bf16<<<1024, 256, 0, stream>>>(x,  x_bf,  (long)BB * SS * DD);
  cvt_to_bf16<<<1024, 256, 0, stream>>>(w1, w1_bf, (long)DD * DFFN);
  cvt_to_bf16<<<1024, 256, 0, stream>>>(w2, w2_bf, (long)DFFN * DD);

  // 2) scores = x . x^T / sqrt(D) + mask   [per batch]; B = x^T is K-contiguous
  gemm_wmma_bf16<0, 1><<<dim3(SS / 64, SS / 128, BB), 128, 0, stream>>>(
      x_bf, x_bf, scores, mask, 0.03125f /* 1/sqrt(1024) */,
      SS, SS, DD, /*ldb=*/DD,
      (long)SS * DD, (long)SS * DD, (long)SS * SS, (long)SS);

  // 3) softmax rows -> bf16 attn
  softmax_rows<<<BB * SS, 256, 0, stream>>>(scores, attn, SS);

  // 4) attn_out = attn . x   [per batch]; B = x row-major
  gemm_wmma_bf16<0, 0><<<dim3(DD / 64, SS / 128, BB), 128, 0, stream>>>(
      attn, x_bf, tmpf, nullptr, 1.0f,
      SS, DD, SS, /*ldb=*/DD,
      (long)SS * SS, (long)SS * DD, (long)SS * DD, 0);

  // 5) h = LN1(x + attn_out)  (+ bf16 copy)
  add_layernorm<<<BB * SS, 256, 0, stream>>>(x, tmpf, g1, be1, h, h_bf);

  // 6) act = GELU(h . w1 + b1)   -> bf16 (aliases scores buffer)
  gemm_wmma_bf16<1, 0><<<dim3(DFFN / 64, (BB * SS) / 128, 1), 128, 0, stream>>>(
      h_bf, w1_bf, act, b1, 1.0f,
      BB * SS, DFFN, DD, /*ldb=*/DFFN,
      0, 0, 0, 0);

  // 7) ffn = act . w2 + b2  -> f32 (aliases attn_out buffer)
  gemm_wmma_bf16<0, 0><<<dim3(DD / 64, (BB * SS) / 128, 1), 128, 0, stream>>>(
      act, w2_bf, tmpf, b2, 1.0f,
      BB * SS, DD, DFFN, /*ldb=*/DD,
      0, 0, 0, 0);

  // 8) out = LN2(h + ffn)
  add_layernorm<<<BB * SS, 256, 0, stream>>>(h, tmpf, g2, be2, out, nullptr);
}
